// SMetaLSTM_52871047413866
// MI455X (gfx1250) — matrix operand order repaired and checked by reading, not compiled, not run on hardware
//
#include <hip/hip_runtime.h>

// SMetaLSTM fused kernel for MI455X (gfx1250, wave32, WMMA).
// One workgroup per node n (500 WGs, 256 threads = 8 waves).
// Phase 1: hypernet (meta -> hx/hh/hb -> Wx, Wh, bg), Wh kept in LDS as bf16.
// Phase 4: 12 LSTM steps. Wave = (m-tile, n-half); each wave computes ALL 4
//          gates for its 16x64 patch, so the LSTM cell runs entirely on WMMA
//          accumulators in registers (no z round-trip through LDS).
// Phase 5: readout head (relu->Wf1->relu->Wf2) and transposed store.

typedef __attribute__((ext_vector_type(16))) __bf16 v16bf;
typedef __attribute__((ext_vector_type(8)))  float  v8f;

#define N_NODES 500
#define BATCH   64
#define HDIM    128
#define EDIM    64
#define LDIM    128
#define TSTEPS  12
#define TOUT    12

// LDS layout (dynamic shared, 161024 B total; CDNA5 allows 320 KB/WG)
#define OFF_WH   0              // 4*128*128 bf16 = 131072  (Wh, [g][ncol][k] K-major)
#define OFF_H    131072         // 64*128    bf16 = 16384   (hidden state, [b][k])
#define OFF_META 147456         // 64  f32
#define OFF_H1   147712         // 3*4*128 f32 (hx, hh, hb)
#define OFF_WX   153856         // 4*128 f32
#define OFF_BGT  155904         // 4*128 f32
#define OFF_XT   157952         // 12*64 f32
#define SMEM_BYTES 161024

__device__ __forceinline__ unsigned short f2bf(float f) {
  union { float f; unsigned u; } v; v.f = f;
  unsigned u = v.u;
  return (unsigned short)((u + 0x7FFFu + ((u >> 16) & 1u)) >> 16); // RNE
}
__device__ __forceinline__ float bf2f(unsigned short h) {
  union { unsigned u; float f; } v; v.u = ((unsigned)h) << 16;
  return v.f;
}

union FragU { uint4 q[2]; v16bf v; };

// A fragment (16x32 bf16, M=m0..m0+15, K=k0..k0+31) from row-major h[b][k].
// Lanes 0-15 hold K 0..15, lanes 16-31 hold K 16..31 (contiguous, matches B).
__device__ __forceinline__ v16bf load_afrag(const unsigned short* sh, int m0, int k0, int lane) {
  int m  = m0 + (lane & 15);
  int kb = k0 + ((lane >> 4) << 4);
  const uint4* p = (const uint4*)(sh + m * HDIM + kb);
  FragU u;
  u.q[0] = p[0];   // K kb..kb+7
  u.q[1] = p[1];   // K kb+8..kb+15
  return u.v;
}

// B fragment (32x16 bf16, K=k0..k0+31, N=n0..n0+15) from Wh stored K-major:
// whg[(ncol)*128 + k]. Lanes 0-15 hold K 0..15 of column n0+lane,
// lanes 16-31 hold K 16..31 -> 16 contiguous bf16 per lane.
__device__ __forceinline__ v16bf load_bfrag(const unsigned short* whg, int n0, int k0, int lane) {
  int j  = n0 + (lane & 15);
  int kb = k0 + ((lane >> 4) << 4);
  const uint4* p = (const uint4*)(whg + j * HDIM + kb);
  FragU u;
  u.q[0] = p[0];
  u.q[1] = p[1];
  return u.v;
}

__global__ void __launch_bounds__(256, 1)
smeta_lstm_node_kernel(const float* __restrict__ x,
                       const float* __restrict__ W1x, const float* __restrict__ b1x,
                       const float* __restrict__ W2x, const float* __restrict__ b2x,
                       const float* __restrict__ W1h, const float* __restrict__ b1h,
                       const float* __restrict__ W2h, const float* __restrict__ b2h,
                       const float* __restrict__ W1b, const float* __restrict__ b1b,
                       const float* __restrict__ W2b, const float* __restrict__ b2b,
                       const float* __restrict__ Wf1, const float* __restrict__ bf1,
                       const float* __restrict__ Wf2, const float* __restrict__ bf2,
                       float* __restrict__ out)
{
  extern __shared__ char smem[];
  unsigned short* s_Wh = (unsigned short*)(smem + OFF_WH);
  unsigned short* s_h  = (unsigned short*)(smem + OFF_H);
  float* s_meta = (float*)(smem + OFF_META);
  float* s_h1   = (float*)(smem + OFF_H1);   // [net][g][l], net 0=x,1=h,2=b
  float* s_Wx   = (float*)(smem + OFF_WX);   // [g][h]
  float* s_bg   = (float*)(smem + OFF_BGT);  // [g][h]
  float* s_xt   = (float*)(smem + OFF_XT);   // [t][b]

  const int n    = blockIdx.x;
  const int tid  = threadIdx.x;
  const int lane = tid & 31;
  const int wave = tid >> 5;

  // ---- Phase 0: meta = x[0,0,n,1:65] -------------------------------------
  if (tid < EDIM) s_meta[tid] = x[(size_t)n * 97 + 1 + tid];
  __syncthreads();

  // ---- Phase 1a: stage-1 hypernet: h? = relu(meta @ W1? + b1?) -----------
  for (int idx = tid; idx < 3 * 4 * LDIM; idx += 256) {
    int net = idx >> 9;
    int r   = idx & 511;                 // g*128 + l
    const float* W1; const float* b1;
    if (net == 0)      { W1 = W1x; b1 = b1x; }
    else if (net == 1) { W1 = W1h; b1 = b1h; }
    else               { W1 = W1b; b1 = b1b; }
    int g = r >> 7, l = r & 127;
    float acc = b1[r];
    for (int e = 0; e < EDIM; ++e)
      acc = fmaf(s_meta[e], W1[(g * EDIM + e) * LDIM + l], acc);
    s_h1[idx] = fmaxf(acc, 0.f);
  }
  __syncthreads();

  // ---- Phase 1b: Wx = hx@W2x + b2x ; bg = hb@W2b + b2b -------------------
  for (int idx = tid; idx < 1024; idx += 256) {
    int half = idx >> 9;
    int r = idx & 511;                   // g*128 + hcol
    int g = r >> 7, hc = r & 127;
    const float* W2   = half ? W2b : W2x;
    const float* b2   = half ? b2b : b2x;
    const float* hsrc = half ? (s_h1 + 2 * 512) : s_h1;
    float acc = b2[r];
    for (int l = 0; l < LDIM; ++l)
      acc = fmaf(hsrc[g * LDIM + l], W2[(g * LDIM + l) * HDIM + hc], acc);
    (half ? s_bg : s_Wx)[r] = acc;
  }

  // ---- Phase 1c: Wh[g] = hh[g]@W2h[g] + b2h[g], stored K-major bf16 ------
  {
    const float* hh = s_h1 + 512;        // net 1
    for (int g = 0; g < 4; ++g) {
      const float* w2g = W2h + (size_t)g * LDIM * (HDIM * HDIM);
      const float* b2g = b2h + (size_t)g * (HDIM * HDIM);
      for (int c = 0; c < 8; ++c) {
        int dbase = c * 2048 + tid;
        __builtin_prefetch(w2g + dbase, 0, 1);   // global_prefetch_b8
        float acc[8];
#pragma unroll
        for (int q = 0; q < 8; ++q) acc[q] = b2g[dbase + q * 256];
        for (int l = 0; l < LDIM; ++l) {
          float s = hh[g * LDIM + l];
          const float* wrow = w2g + (size_t)l * (HDIM * HDIM) + dbase;
#pragma unroll
          for (int q = 0; q < 8; ++q)
            acc[q] = fmaf(s, wrow[q * 256], acc[q]);
        }
#pragma unroll
        for (int q = 0; q < 8; ++q) {
          int d = dbase + q * 256;
          int i = d >> 7, j = d & 127;   // Wh[g][i][j]; store transposed [j][i]
          s_Wh[((g * HDIM) + j) * HDIM + i] = f2bf(acc[q]);
        }
      }
    }
  }

  // ---- Phase 2/3: stage xt, zero h --------------------------------------
  for (int idx = tid; idx < TSTEPS * BATCH; idx += 256) {
    int t = idx >> 6, b = idx & 63;
    s_xt[idx] = x[((size_t)(b * TSTEPS + t) * N_NODES + n) * 97];
  }
  for (int idx = tid; idx < BATCH * HDIM; idx += 256) s_h[idx] = 0;
  __syncthreads();

  // ---- Phase 4: 12 LSTM steps --------------------------------------------
  // Wave owns output patch: rows [m0, m0+16), cols [nbase, nbase+64), all 4 gates.
  const int m0      = (wave >> 1) * 16;
  const int nbase   = (wave & 1) * 64;
  const int colLane = lane & 15;
  const int mOfs    = (lane >> 4) << 3;

  // Per-lane gate parameters, hoisted out of the time loop.
  float wxv[4][4], bgv[4][4];
#pragma unroll
  for (int g = 0; g < 4; ++g)
#pragma unroll
    for (int nt = 0; nt < 4; ++nt) {
      int col = nbase + nt * 16 + colLane;
      wxv[g][nt] = s_Wx[g * HDIM + col];
      bgv[g][nt] = s_bg[g * HDIM + col];
    }

  float creg[4][8];
#pragma unroll
  for (int nt = 0; nt < 4; ++nt)
#pragma unroll
    for (int r = 0; r < 8; ++r) creg[nt][r] = 0.f;

  for (int t = 0; t < TSTEPS; ++t) {
    v8f acc[4][4];                       // [gate][ntile], 128 VGPRs
#pragma unroll
    for (int g = 0; g < 4; ++g)
#pragma unroll
      for (int nt = 0; nt < 4; ++nt) acc[g][nt] = (v8f){};

    for (int kt = 0; kt < 4; ++kt) {
      int k0 = kt * 32;
      v16bf afr = load_afrag(s_h, m0, k0, lane);   // reused by 16 WMMAs
#pragma unroll
      for (int g = 0; g < 4; ++g) {
        const unsigned short* whg = s_Wh + g * HDIM * HDIM;
#pragma unroll
        for (int nt = 0; nt < 4; ++nt) {
          v16bf bfr = load_bfrag(whg, nbase + nt * 16, k0, lane);
          acc[g][nt] = __builtin_amdgcn_wmma_f32_16x16x32_bf16(
              false, afr, false, bfr, (short)0, acc[g][nt], false, false);
        }
      }
    }
    // All s_h reads are dscnt-complete before the last WMMA issues; after this
    // barrier every wave is done reading h(t-1) and may overwrite it.
    __syncthreads();

    // LSTM cell entirely in registers: all 4 gates of element (m, col) live
    // in the same lane / same accumulator slot r.
    const float* xtcol = s_xt + t * BATCH;
#pragma unroll
    for (int r = 0; r < 8; ++r) {
      int m = m0 + mOfs + r;
      float xv = xtcol[m];
#pragma unroll
      for (int nt = 0; nt < 4; ++nt) {
        float z0 = acc[0][nt][r] + xv * wxv[0][nt] + bgv[0][nt];
        float z1 = acc[1][nt][r] + xv * wxv[1][nt] + bgv[1][nt];
        float z2 = acc[2][nt][r] + xv * wxv[2][nt] + bgv[2][nt];
        float z3 = acc[3][nt][r] + xv * wxv[3][nt] + bgv[3][nt];
        float gv = tanhf(z0);
        float iv = 1.f / (1.f + __expf(-z1));
        float fv = 1.f / (1.f + __expf(-z2));
        float ov = 1.f / (1.f + __expf(-z3));
        float cv = fmaf(gv, iv, creg[nt][r] * fv);
        creg[nt][r] = cv;
        s_h[m * HDIM + nbase + nt * 16 + colLane] = f2bf(tanhf(cv) * ov);
      }
    }
    __syncthreads();
  }

  // ---- Phase 5: readout head ---------------------------------------------
  float* s_hid = (float*)(smem + OFF_WH);  // reuse dead Wh region (needs 16 KB)
#pragma unroll
  for (int q = 0; q < 16; ++q) {
    int o = q * 256 + tid;                 // 4096 outputs (64 x 64)
    int b = o >> 6, jc = o & 63;
    float acc = bf1[jc];
    for (int k = 0; k < HDIM; ++k)
      acc = fmaf(fmaxf(bf2f(s_h[b * HDIM + k]), 0.f), Wf1[k * 64 + jc], acc);
    s_hid[o] = fmaxf(acc, 0.f);
  }
  __syncthreads();
  for (int o = tid; o < BATCH * TOUT; o += 256) {
    int b = o / TOUT, t = o % TOUT;
    float acc = bf2[t];
    for (int k = 0; k < 64; ++k)
      acc = fmaf(s_hid[b * 64 + k], Wf2[k * TOUT + t], acc);
    out[(size_t)(b * TOUT + t) * N_NODES + n] = acc;   // (B, T_OUT, N, 1)
  }
}

extern "C" void kernel_launch(void* const* d_in, const int* in_sizes, int n_in,
                              void* d_out, int out_size, void* d_ws, size_t ws_size,
                              hipStream_t stream) {
  (void)in_sizes; (void)n_in; (void)d_ws; (void)ws_size; (void)out_size;
  const float* x   = (const float*)d_in[0];
  const float* W1x = (const float*)d_in[1];
  const float* b1x = (const float*)d_in[2];
  const float* W2x = (const float*)d_in[3];
  const float* b2x = (const float*)d_in[4];
  const float* W1h = (const float*)d_in[5];
  const float* b1h = (const float*)d_in[6];
  const float* W2h = (const float*)d_in[7];
  const float* b2h = (const float*)d_in[8];
  const float* W1b = (const float*)d_in[9];
  const float* b1b = (const float*)d_in[10];
  const float* W2b = (const float*)d_in[11];
  const float* b2b = (const float*)d_in[12];
  const float* Wf1 = (const float*)d_in[13];
  const float* bf1 = (const float*)d_in[14];
  const float* Wf2 = (const float*)d_in[15];
  const float* bf2 = (const float*)d_in[16];

  // Opt in to >64KB dynamic LDS (no-op if already allowed; safe under capture).
  (void)hipFuncSetAttribute((const void*)smeta_lstm_node_kernel,
                            hipFuncAttributeMaxDynamicSharedMemorySize, SMEM_BYTES);

  smeta_lstm_node_kernel<<<N_NODES, 256, SMEM_BYTES, stream>>>(
      x, W1x, b1x, W2x, b2x, W1h, b1h, W2h, b2h,
      W1b, b1b, W2b, b2b, Wf1, bf1, Wf2, bf2, (float*)d_out);
}
